// TransformerSeqLayer_40106404610681
// MI455X (gfx1250) — compile-verified, matrix operand order checked by reading, not run on hardware
//
#include <hip/hip_runtime.h>

// =====================================================================
// MI455X (gfx1250, wave32) implementation of the TransformerSeqLayer.
// All large GEMMs run on v_wmma_f32_16x16x32_bf16 (bf16 in, f32 acc).
// A-tiles staged to LDS via GLOBAL_LOAD_ASYNC_TO_LDS_B128; B-tiles
// stored transposed+padded in LDS so every WMMA fragment is two
// ds_load_b128 (no scalar u16 LDS traffic).
// MoE exploits top-2 sparsity via compacted grouped GEMMs.
// Workspace requirement: ~310 MB.
// =====================================================================

typedef __attribute__((ext_vector_type(16))) __bf16 v16bf;
typedef __attribute__((ext_vector_type(8)))  __bf16 v8bf;
typedef __attribute__((ext_vector_type(8)))  float  v8f;
typedef __attribute__((ext_vector_type(4)))  int    v4i;

#define HD   1024   // hidden
#define NH   8      // heads
#define DH_  128    // head dim
#define LSP  512    // attn span
#define MM   512    // query len
#define BB   8      // batch
#define NEXP 16
#define DFF  2048
#define NTOK (BB*MM)          // 4096
#define NSLOT (2*NTOK)        // 8192 assignments (top-2, always exactly 2)
#define SLOTCAP 8448

#define LDA 40      // padded LDS row stride (elements) for 32-wide K tiles
#define LDP 552     // padded LDS row stride for the attention prob band

// -------------------------------------------------- async LDS staging
#if defined(__has_builtin)
#if __has_builtin(__builtin_amdgcn_global_load_async_to_lds_b128)
#define HAVE_ASYNC 1
#endif
#endif

// From the toolchain diagnostic: the async builtin takes
// (v4i addrspace(1)*, v4i addrspace(3)*, imm offset, imm cpol).
#define GPTR(p) ((__attribute__((address_space(1))) v4i*)(p))
#define LPTR(p) ((__attribute__((address_space(3))) v4i*)(p))

__device__ __forceinline__ void async_wait0() {
#if defined(HAVE_ASYNC)
#if __has_builtin(__builtin_amdgcn_s_wait_asynccnt)
  __builtin_amdgcn_s_wait_asynccnt(0);
#else
  asm volatile("s_wait_asynccnt 0x0" ::: "memory");
#endif
#endif
}

// ---------------------------------------------------------------- WMMA
__device__ __forceinline__ v8f wmma_bf16(v16bf a, v16bf b, v8f c) {
  return __builtin_amdgcn_wmma_f32_16x16x32_bf16(false, a, false, b,
                                                 (short)0, c, false, false);
}

__device__ __forceinline__ v16bf frag_cat(v8bf lo, v8bf hi) {
  return __builtin_shufflevector(lo, hi, 0, 1, 2, 3, 4, 5, 6, 7,
                                 8, 9, 10, 11, 12, 13, 14, 15);
}

// A fragment: 16x32 bf16, row-major, row stride ld (elements).
// Per lane: two contiguous 16B chunks -> two b128 loads.
__device__ __forceinline__ v16bf frag_a(const __bf16* __restrict__ s, int ld) {
  int lane = threadIdx.x & 31;
  const __bf16* row = s + (size_t)(lane & 15) * ld + (lane >> 4) * 8;
  return frag_cat(*(const v8bf*)row, *(const v8bf*)(row + 16));
}

// B fragment from K-contiguous storage (transposed LDS tile or K^T in
// global): p points at this lane's column, elements contiguous in K.
__device__ __forceinline__ v16bf frag_b_t(const __bf16* __restrict__ p) {
  int h8 = ((threadIdx.x & 31) >> 4) * 8;
  return frag_cat(*(const v8bf*)(p + h8), *(const v8bf*)(p + 16 + h8));
}

// B fragment: 32x16 bf16 row-major (strided; only used for small kpe GEMM)
__device__ __forceinline__ v16bf frag_b(const __bf16* __restrict__ s, int ld) {
  int lane = threadIdx.x & 31;
  int n = lane & 15;
  int h8 = (lane >> 4) * 8;
  v16bf b;
#pragma unroll
  for (int i = 0; i < 8; ++i) b[i] = s[(size_t)(h8 + i) * ld + n];
#pragma unroll
  for (int i = 0; i < 8; ++i) b[8 + i] = s[(size_t)(16 + h8 + i) * ld + n];
  return b;
}

// B fragment row-major with row guard (rows >= rowmax read as 0).
__device__ __forceinline__ v16bf frag_b_guard(const __bf16* __restrict__ colbase,
                                              size_t ld, int row0, int rowmax) {
  int h8 = ((threadIdx.x & 31) >> 4) * 8;
  v16bf b;
#pragma unroll
  for (int i = 0; i < 8; ++i) {
    int r = row0 + h8 + i;
    b[i] = (r < rowmax) ? colbase[(size_t)r * ld] : (__bf16)0.0f;
  }
#pragma unroll
  for (int i = 0; i < 8; ++i) {
    int r = row0 + 16 + h8 + i;
    b[8 + i] = (r < rowmax) ? colbase[(size_t)r * ld] : (__bf16)0.0f;
  }
  return b;
}

__device__ __forceinline__ void frag_c_store(float* __restrict__ d, int ld, v8f c) {
  int lane = threadIdx.x & 31;
  int n = lane & 15;
  int h = lane >> 4;
#pragma unroll
  for (int r = 0; r < 8; ++r) d[(size_t)(8 * h + r) * ld + n] = c[r];
}

// Stage one 32xN-tile of row-major B into transposed padded LDS:
// BsT[n][k], row stride LDA. Thread t handles column n = t&63,
// K-octet kh = t>>6; per-i reads are coalesced across lanes.
__device__ __forceinline__ void stage_bT(const __bf16* __restrict__ B, int N,
                                         int kk, int n0, __bf16* __restrict__ BsT) {
  int t = threadIdx.x;
  int nn = t & 63, kh = t >> 6;
  __bf16 tmp[8];
#pragma unroll
  for (int i = 0; i < 8; ++i)
    tmp[i] = B[(size_t)(kk + kh * 8 + i) * N + n0 + nn];
  *(v8bf*)(BsT + nn * LDA + kh * 8) = *(v8bf*)tmp;
}

// ------------------------------------------------------------- convert
__global__ void k_cvt(const float* __restrict__ s, __bf16* __restrict__ d, size_t n) {
  size_t i = (size_t)blockIdx.x * blockDim.x + threadIdx.x;
  size_t st = (size_t)gridDim.x * blockDim.x;
  for (; i < n; i += st) d[i] = (__bf16)s[i];
}

// hall[b, 0..511] = cache, hall[b, 512..1023] = h  (bf16)
__global__ void k_build_hall(const float* __restrict__ hc, const float* __restrict__ h,
                             __bf16* __restrict__ hall) {
  size_t n = (size_t)BB * 1024 * HD;
  size_t st = (size_t)gridDim.x * blockDim.x;
  for (size_t i = (size_t)blockIdx.x * blockDim.x + threadIdx.x; i < n; i += st) {
    size_t col = i & (HD - 1);
    size_t row = i >> 10;
    size_t b = row >> 10;
    size_t p = row & 1023;
    float v = (p < LSP) ? hc[(b * LSP + p) * HD + col]
                        : h[(b * MM + (p - LSP)) * HD + col];
    hall[i] = (__bf16)v;
  }
}

// --------------------------------------------------------------- GEMM
// C[M,N] = A[M,K] * B[K,N]   (bf16 x bf16 -> f32 acc)
// amode 1: logical row r maps to hall row (r/512)*1024 + 512 + r%512
// emode 0: f32 out,  emode 1: bf16 out (scaled)
__global__ __launch_bounds__(256) void k_gemm(
    const __bf16* __restrict__ A, const __bf16* __restrict__ B,
    int M, int N, int Kd, int amode, int emode, float scale,
    float* __restrict__ Cf, __bf16* __restrict__ Cb) {
  __shared__ __bf16 As[64 * LDA];
  __shared__ __bf16 BsT[64 * LDA];
  int tid = threadIdx.x, wave = tid >> 5, lane = tid & 31;
  int n0 = blockIdx.x * 64, m0 = blockIdx.y * 64;
  int mi = wave & 3, nj = (wave >> 2) * 2;
  v8f c0 = {}, c1 = {};
  int ar = tid >> 2, aseg = (tid & 3) * 8;
  int R = m0 + ar;
  size_t arow = (amode == 1) ? (size_t)((R >> 9) * 1024 + 512 + (R & 511))
                             : (size_t)R;
  const __bf16* aptr = A + arow * (size_t)Kd + aseg;
  const __bf16* bt0 = BsT + (size_t)(nj * 16 + (lane & 15)) * LDA;
  const __bf16* bt1 = bt0 + 16 * LDA;
  for (int kk = 0; kk < Kd; kk += 32) {
#if defined(HAVE_ASYNC)
    __builtin_amdgcn_global_load_async_to_lds_b128(
        GPTR(aptr + kk), LPTR(As + ar * LDA + aseg), 0, 0);
#else
    *(v8bf*)(As + ar * LDA + aseg) = *(const v8bf*)(aptr + kk);
#endif
    stage_bT(B, N, kk, n0, BsT);
    if (kk + 32 < Kd) __builtin_prefetch(aptr + kk + 32, 0, 0);
    async_wait0();
    __syncthreads();
    v16bf a = frag_a(As + mi * 16 * LDA, LDA);
    c0 = wmma_bf16(a, frag_b_t(bt0), c0);
    c1 = wmma_bf16(a, frag_b_t(bt1), c1);
    __syncthreads();
  }
  int n = lane & 15, h = lane >> 4;
#pragma unroll
  for (int w = 0; w < 2; ++w) {
    v8f& c = w ? c1 : c0;
    int col = n0 + (nj + w) * 16 + n;
#pragma unroll
    for (int r = 0; r < 8; ++r) {
      int row = m0 + mi * 16 + 8 * h + r;
      float v = c[r] * scale;
      if (emode == 0) Cf[(size_t)row * N + col] = v;
      else            Cb[(size_t)row * N + col] = (__bf16)v;
    }
  }
}

// ----------------------------------------------------------- attention
// One block per (head-batch bk, 16-query tile). Banded score GEMM
// (528-wide), pos-bias scatter-add, softmax, PV GEMM.
__global__ __launch_bounds__(256) void k_attn(
    const __bf16* __restrict__ qb, const __bf16* __restrict__ kb,
    const __bf16* __restrict__ vb, const __bf16* __restrict__ kpe,
    __bf16* __restrict__ ctx) {
  __shared__ float Sqk[16][528];
  __shared__ __bf16 Pbuf[16][LDP];
  int bk = blockIdx.x;            // 0..63
  int m0 = blockIdx.y * 16;       // query tile
  int b = bk >> 3, head = bk & 7;
  int tid = threadIdx.x, wave = tid >> 5, lane = tid & 31;

  // zero the padded probability band (edges must be exact zeros)
  for (int i = tid; i < 16 * LDP / 2; i += 256) ((unsigned int*)Pbuf)[i] = 0u;

  const __bf16* qrow0 = qb + ((size_t)(b * MM + m0)) * HD + head * DH_;

  // ---- content scores: 33 key tiles, n = m0 + j, j in [0,528)
  for (int t = wave; t < 33; t += 8) {
    v8f acc = {};
    int n0 = m0 + t * 16;
#pragma unroll
    for (int kc = 0; kc < 4; ++kc) {
      v16bf a = frag_a(qrow0 + kc * 32, HD);
      const __bf16* kp = kb + ((size_t)(b * 1024 + n0 + (lane & 15))) * HD
                            + head * DH_ + kc * 32;   // column-contiguous K^T
      acc = wmma_bf16(a, frag_b_t(kp), acc);
    }
    frag_c_store(&Sqk[0][0] + t * 16, 528, acc);
  }
  __syncthreads();

  // ---- positional scores: scatter-add at skewed offset j = m + l
  for (int t = wave; t < 32; t += 8) {
    v8f acc = {};
    int l0 = t * 16;
#pragma unroll
    for (int kc = 0; kc < 4; ++kc) {
      v16bf a = frag_a(qrow0 + kc * 32, HD);
      acc = wmma_bf16(a, frag_b(kpe + (size_t)(kc * 32) * LSP + l0, LSP), acc);
    }
    int n = lane & 15, h = lane >> 4;
#pragma unroll
    for (int r = 0; r < 8; ++r) {
      int m = 8 * h + r;
      Sqk[m][m + l0 + n] += acc[r];
    }
  }
  __syncthreads();

  // ---- softmax over l in [0,512): each wave handles two rows
#pragma unroll
  for (int rr = 0; rr < 2; ++rr) {
    int m = wave * 2 + rr;
    float mx = -1e30f;
    for (int l = lane; l < LSP; l += 32) mx = fmaxf(mx, Sqk[m][m + l]);
#pragma unroll
    for (int o = 16; o; o >>= 1) mx = fmaxf(mx, __shfl_xor(mx, o, 32));
    float sum = 0.f;
    for (int l = lane; l < LSP; l += 32) {
      float e = __expf(Sqk[m][m + l] - mx);
      Sqk[m][m + l] = e;
      sum += e;
    }
#pragma unroll
    for (int o = 16; o; o >>= 1) sum += __shfl_xor(sum, o, 32);
    float inv = 1.f / sum;
    for (int l = lane; l < LSP; l += 32)
      Pbuf[m][m + l] = (__bf16)(Sqk[m][m + l] * inv);
  }
  __syncthreads();

  // ---- PV: each wave owns one 16-wide d-tile; 17 K=32 chunks over band
  {
    v8f acc = {};
    int d0 = head * DH_ + wave * 16;
    const __bf16* vcol = vb + (size_t)b * 1024 * HD + d0 + (lane & 15);
    for (int kc = 0; kc < 17; ++kc) {
      v16bf a = frag_a(&Pbuf[0][0] + kc * 32, LDP);
      acc = wmma_bf16(a, frag_b_guard(vcol, HD, m0 + kc * 32, 1024), acc);
    }
    int n = lane & 15, h = lane >> 4;
#pragma unroll
    for (int r = 0; r < 8; ++r) {
      int m = 8 * h + r;
      ctx[((size_t)(b * MM + m0 + m)) * HD + d0 + n] = (__bf16)acc[r];
    }
  }
}

// ------------------------------------------------------------ layernorm
__global__ __launch_bounds__(256) void k_ln1(
    const float* __restrict__ h, const float* __restrict__ attn,
    const float* __restrict__ g, const float* __restrict__ bta,
    float* __restrict__ h1f, __bf16* __restrict__ h1b) {
  int t = blockIdx.x;
  const float* xa = h + (size_t)t * HD;
  const float* xb = attn + (size_t)t * HD;
  __shared__ float red[8];
  float loc[4], s = 0.f;
#pragma unroll
  for (int i = 0; i < 4; ++i) {
    int idx = threadIdx.x + i * 256;
    loc[i] = xa[idx] + xb[idx];
    s += loc[i];
  }
#pragma unroll
  for (int o = 16; o; o >>= 1) s += __shfl_xor(s, o, 32);
  if ((threadIdx.x & 31) == 0) red[threadIdx.x >> 5] = s;
  __syncthreads();
  float tot = 0.f;
#pragma unroll
  for (int w = 0; w < 8; ++w) tot += red[w];
  float mu = tot * (1.f / HD);
  float vs = 0.f;
#pragma unroll
  for (int i = 0; i < 4; ++i) { float d = loc[i] - mu; vs += d * d; }
#pragma unroll
  for (int o = 16; o; o >>= 1) vs += __shfl_xor(vs, o, 32);
  __syncthreads();
  if ((threadIdx.x & 31) == 0) red[threadIdx.x >> 5] = vs;
  __syncthreads();
  float vt = 0.f;
#pragma unroll
  for (int w = 0; w < 8; ++w) vt += red[w];
  float inv = rsqrtf(vt * (1.f / HD) + 1e-5f);
#pragma unroll
  for (int i = 0; i < 4; ++i) {
    int idx = threadIdx.x + i * 256;
    float y = (loc[i] - mu) * inv * g[idx] + bta[idx];
    h1f[(size_t)t * HD + idx] = y;
    h1b[(size_t)t * HD + idx] = (__bf16)y;
  }
}

// ----------------------------------------------------------------- MoE
__global__ void k_zero_counts(int* __restrict__ count) {
  if (threadIdx.x < NEXP) count[threadIdx.x] = 0;
}

__global__ void k_gate(const float* __restrict__ h1, const float* __restrict__ gw,
                       const float* __restrict__ gb, int* __restrict__ topi,
                       float* __restrict__ topw, int* __restrict__ count) {
  int t = blockIdx.x;
  int lane = threadIdx.x;
  __shared__ float logits[NEXP];
  if (lane < NEXP) {
    float s = gb[lane];
    const float* x = h1 + (size_t)t * HD;
    for (int i = 0; i < HD; ++i) s += x[i] * gw[i * NEXP + lane];
    logits[lane] = s;
  }
  __syncthreads();
  if (lane == 0) {
    int i0 = 0; float v0 = logits[0];
    for (int e = 1; e < NEXP; ++e) if (logits[e] > v0) { v0 = logits[e]; i0 = e; }
    int i1 = (i0 == 0) ? 1 : 0; float v1 = logits[i1];
    for (int e = 0; e < NEXP; ++e)
      if (e != i0 && logits[e] > v1) { v1 = logits[e]; i1 = e; }
    float e0 = 1.0f, e1 = __expf(v1 - v0);   // stable 2-way softmax
    float inv = 1.f / (e0 + e1);
    topi[2 * t] = i0; topi[2 * t + 1] = i1;
    topw[2 * t] = e0 * inv; topw[2 * t + 1] = e1 * inv;
    atomicAdd(&count[i0], 1);
    atomicAdd(&count[i1], 1);
  }
}

__global__ void k_scan(const int* __restrict__ count, int* __restrict__ base,
                       int* __restrict__ fill) {
  if (threadIdx.x == 0) {
    int acc = 0;
    for (int e = 0; e < NEXP; ++e) { base[e] = acc; acc += count[e]; }
  }
  if (threadIdx.x < NEXP) fill[threadIdx.x] = 0;
}

__global__ void k_assign(const int* __restrict__ topi, const int* __restrict__ base,
                         int* __restrict__ fill, int* __restrict__ slot_token,
                         int* __restrict__ inv_slot) {
  int t = blockIdx.x * blockDim.x + threadIdx.x;
  if (t >= NTOK) return;
#pragma unroll
  for (int j = 0; j < 2; ++j) {
    int e = topi[2 * t + j];
    int pos = atomicAdd(&fill[e], 1);
    int slot = base[e] + pos;
    slot_token[slot] = t;
    inv_slot[2 * t + j] = slot;
  }
}

// y1[slot] = relu(h1[token] @ w1[e] + b1[e])   (bf16 out)
__global__ __launch_bounds__(256) void k_moe1(
    const __bf16* __restrict__ h1b, const __bf16* __restrict__ w1b,
    const float* __restrict__ b1, const int* __restrict__ base,
    const int* __restrict__ count, const int* __restrict__ slot_token,
    __bf16* __restrict__ y1) {
  int e = blockIdx.y;
  int limit = base[e] + count[e];
  int m0 = base[e] + blockIdx.z * 64;
  if (m0 >= limit) return;
  int n0 = blockIdx.x * 64;
  const __bf16* B = w1b + (size_t)e * HD * DFF;
  __shared__ __bf16 As[64 * LDA];
  __shared__ __bf16 BsT[64 * LDA];
  __shared__ int tok[64];
  int tid = threadIdx.x, wave = tid >> 5, lane = tid & 31;
  if (tid < 64) {
    int slot = m0 + tid;
    tok[tid] = (slot < NSLOT) ? slot_token[slot] : 0;
  }
  __syncthreads();
  int mi = wave & 3, nj = (wave >> 2) * 2;
  v8f c0 = {}, c1 = {};
  int ar = tid >> 2, aseg = (tid & 3) * 8;
  const __bf16* aptr = h1b + (size_t)tok[ar] * HD + aseg;
  const __bf16* bt0 = BsT + (size_t)(nj * 16 + (lane & 15)) * LDA;
  const __bf16* bt1 = bt0 + 16 * LDA;
  for (int kk = 0; kk < HD; kk += 32) {
#if defined(HAVE_ASYNC)
    __builtin_amdgcn_global_load_async_to_lds_b128(
        GPTR(aptr + kk), LPTR(As + ar * LDA + aseg), 0, 0);
#else
    *(v8bf*)(As + ar * LDA + aseg) = *(const v8bf*)(aptr + kk);
#endif
    stage_bT(B, DFF, kk, n0, BsT);
    async_wait0();
    __syncthreads();
    v16bf a = frag_a(As + mi * 16 * LDA, LDA);
    c0 = wmma_bf16(a, frag_b_t(bt0), c0);
    c1 = wmma_bf16(a, frag_b_t(bt1), c1);
    __syncthreads();
  }
  int n = lane & 15, h = lane >> 4;
  const float* be = b1 + (size_t)e * DFF;
#pragma unroll
  for (int w = 0; w < 2; ++w) {
    v8f& c = w ? c1 : c0;
    int col = n0 + (nj + w) * 16 + n;
#pragma unroll
    for (int r = 0; r < 8; ++r) {
      int slot = m0 + mi * 16 + 8 * h + r;
      if (slot < limit)
        y1[(size_t)slot * DFF + col] = (__bf16)fmaxf(c[r] + be[col], 0.f);
    }
  }
}

// y2[slot] = y1[slot] @ w2[e] + b2[e]   (f32 out)
__global__ __launch_bounds__(256) void k_moe2(
    const __bf16* __restrict__ y1, const __bf16* __restrict__ w2b,
    const float* __restrict__ b2, const int* __restrict__ base,
    const int* __restrict__ count, float* __restrict__ y2) {
  int e = blockIdx.y;
  int limit = base[e] + count[e];
  int m0 = base[e] + blockIdx.z * 64;
  if (m0 >= limit) return;
  int n0 = blockIdx.x * 64;
  const __bf16* B = w2b + (size_t)e * DFF * HD;
  __shared__ __bf16 As[64 * LDA];
  __shared__ __bf16 BsT[64 * LDA];
  int tid = threadIdx.x, wave = tid >> 5, lane = tid & 31;
  int mi = wave & 3, nj = (wave >> 2) * 2;
  v8f c0 = {}, c1 = {};
  int ar = tid >> 2, aseg = (tid & 3) * 8;
  const __bf16* aptr = y1 + (size_t)(m0 + ar) * DFF + aseg;
  const __bf16* bt0 = BsT + (size_t)(nj * 16 + (lane & 15)) * LDA;
  const __bf16* bt1 = bt0 + 16 * LDA;
  for (int kk = 0; kk < DFF; kk += 32) {
#if defined(HAVE_ASYNC)
    __builtin_amdgcn_global_load_async_to_lds_b128(
        GPTR(aptr + kk), LPTR(As + ar * LDA + aseg), 0, 0);
#else
    *(v8bf*)(As + ar * LDA + aseg) = *(const v8bf*)(aptr + kk);
#endif
    stage_bT(B, HD, kk, n0, BsT);
    async_wait0();
    __syncthreads();
    v16bf a = frag_a(As + mi * 16 * LDA, LDA);
    c0 = wmma_bf16(a, frag_b_t(bt0), c0);
    c1 = wmma_bf16(a, frag_b_t(bt1), c1);
    __syncthreads();
  }
  int n = lane & 15, h = lane >> 4;
  const float* be = b2 + (size_t)e * HD;
#pragma unroll
  for (int w = 0; w < 2; ++w) {
    v8f& c = w ? c1 : c0;
    int col = n0 + (nj + w) * 16 + n;
#pragma unroll
    for (int r = 0; r < 8; ++r) {
      int slot = m0 + mi * 16 + 8 * h + r;
      if (slot < limit) y2[(size_t)slot * HD + col] = c[r] + be[col];
    }
  }
}

// out = LN(h1 + w0*y2[s0] + w1*y2[s1])
__global__ __launch_bounds__(256) void k_ln2(
    const float* __restrict__ h1f, const float* __restrict__ y2,
    const int* __restrict__ inv_slot, const float* __restrict__ topw,
    const float* __restrict__ g, const float* __restrict__ bta,
    float* __restrict__ out) {
  int t = blockIdx.x;
  int s0 = inv_slot[2 * t], s1 = inv_slot[2 * t + 1];
  float w0 = topw[2 * t], w1 = topw[2 * t + 1];
  const float* xa = h1f + (size_t)t * HD;
  const float* ya = y2 + (size_t)s0 * HD;
  const float* yb = y2 + (size_t)s1 * HD;
  __shared__ float red[8];
  float loc[4], s = 0.f;
#pragma unroll
  for (int i = 0; i < 4; ++i) {
    int idx = threadIdx.x + i * 256;
    loc[i] = xa[idx] + w0 * ya[idx] + w1 * yb[idx];
    s += loc[i];
  }
#pragma unroll
  for (int o = 16; o; o >>= 1) s += __shfl_xor(s, o, 32);
  if ((threadIdx.x & 31) == 0) red[threadIdx.x >> 5] = s;
  __syncthreads();
  float tot = 0.f;
#pragma unroll
  for (int w = 0; w < 8; ++w) tot += red[w];
  float mu = tot * (1.f / HD);
  float vs = 0.f;
#pragma unroll
  for (int i = 0; i < 4; ++i) { float d = loc[i] - mu; vs += d * d; }
#pragma unroll
  for (int o = 16; o; o >>= 1) vs += __shfl_xor(vs, o, 32);
  __syncthreads();
  if ((threadIdx.x & 31) == 0) red[threadIdx.x >> 5] = vs;
  __syncthreads();
  float vt = 0.f;
#pragma unroll
  for (int w = 0; w < 8; ++w) vt += red[w];
  float inv = rsqrtf(vt * (1.f / HD) + 1e-5f);
#pragma unroll
  for (int i = 0; i < 4; ++i) {
    int idx = threadIdx.x + i * 256;
    out[(size_t)t * HD + idx] = (loc[i] - mu) * inv * g[idx] + bta[idx];
  }
}

// ================================================================ host
static char* ws_take(char*& p, size_t bytes) {
  char* r = p;
  p += (bytes + 255) & ~(size_t)255;
  return r;
}

extern "C" void kernel_launch(void* const* d_in, const int* in_sizes, int n_in,
                              void* d_out, int out_size, void* d_ws, size_t ws_size,
                              hipStream_t stream) {
  (void)in_sizes; (void)n_in; (void)out_size; (void)ws_size;
  const float* h      = (const float*)d_in[0];
  const float* hcache = (const float*)d_in[1];
  const float* key_pe = (const float*)d_in[2];
  const float* Wq     = (const float*)d_in[3];
  const float* Wk     = (const float*)d_in[4];
  const float* Wv     = (const float*)d_in[5];
  const float* Wo     = (const float*)d_in[6];
  const float* ln1_g  = (const float*)d_in[7];
  const float* ln1_b  = (const float*)d_in[8];
  const float* gate_w = (const float*)d_in[9];
  const float* gate_b = (const float*)d_in[10];
  const float* w1     = (const float*)d_in[11];
  const float* b1     = (const float*)d_in[12];
  const float* w2     = (const float*)d_in[13];
  const float* b2     = (const float*)d_in[14];
  const float* ln2_g  = (const float*)d_in[15];
  const float* ln2_b  = (const float*)d_in[16];
  float* out = (float*)d_out;

  char* p = (char*)d_ws;
  __bf16* hall  = (__bf16*)ws_take(p, (size_t)BB * 1024 * HD * 2);
  __bf16* wqb   = (__bf16*)ws_take(p, (size_t)HD * HD * 2);
  __bf16* wkb   = (__bf16*)ws_take(p, (size_t)HD * HD * 2);
  __bf16* wvb   = (__bf16*)ws_take(p, (size_t)HD * HD * 2);
  __bf16* wob   = (__bf16*)ws_take(p, (size_t)HD * HD * 2);
  __bf16* kpeb  = (__bf16*)ws_take(p, (size_t)DH_ * LSP * 2);
  __bf16* qb    = (__bf16*)ws_take(p, (size_t)NTOK * HD * 2);
  __bf16* kbuf  = (__bf16*)ws_take(p, (size_t)BB * 1024 * HD * 2);
  __bf16* vbuf  = (__bf16*)ws_take(p, (size_t)BB * 1024 * HD * 2);
  __bf16* ctxb  = (__bf16*)ws_take(p, (size_t)NTOK * HD * 2);
  float*  attno = (float*)ws_take(p, (size_t)NTOK * HD * 4);
  float*  h1f   = (float*)ws_take(p, (size_t)NTOK * HD * 4);
  __bf16* h1b   = (__bf16*)ws_take(p, (size_t)NTOK * HD * 2);
  __bf16* w1b   = (__bf16*)ws_take(p, (size_t)NEXP * HD * DFF * 2);
  __bf16* w2b   = (__bf16*)ws_take(p, (size_t)NEXP * DFF * HD * 2);
  int*    topi  = (int*)ws_take(p, (size_t)NSLOT * 4);
  float*  topw  = (float*)ws_take(p, (size_t)NSLOT * 4);
  int*    count = (int*)ws_take(p, 64);
  int*    base  = (int*)ws_take(p, 64);
  int*    fill  = (int*)ws_take(p, 64);
  int*    slot_token = (int*)ws_take(p, (size_t)SLOTCAP * 4);
  int*    inv_slot   = (int*)ws_take(p, (size_t)NSLOT * 4);
  __bf16* y1    = (__bf16*)ws_take(p, (size_t)SLOTCAP * DFF * 2);
  float*  y2    = (float*)ws_take(p, (size_t)SLOTCAP * HD * 4);

  // ---- convert to bf16
  k_build_hall<<<2048, 256, 0, stream>>>(hcache, h, hall);
  k_cvt<<<1024, 256, 0, stream>>>(Wq, wqb, (size_t)HD * HD);
  k_cvt<<<1024, 256, 0, stream>>>(Wk, wkb, (size_t)HD * HD);
  k_cvt<<<1024, 256, 0, stream>>>(Wv, wvb, (size_t)HD * HD);
  k_cvt<<<1024, 256, 0, stream>>>(Wo, wob, (size_t)HD * HD);
  k_cvt<<<64, 256, 0, stream>>>(key_pe, kpeb, (size_t)DH_ * LSP);
  k_cvt<<<4096, 256, 0, stream>>>(w1, w1b, (size_t)NEXP * HD * DFF);
  k_cvt<<<4096, 256, 0, stream>>>(w2, w2b, (size_t)NEXP * DFF * HD);

  // ---- QKV projections (q pre-scaled by 1/sqrt(D))
  k_gemm<<<dim3(16, 64), 256, 0, stream>>>(hall, wqb, NTOK, HD, HD, 1, 1,
                                           0.08838834764831845f, nullptr, qb);
  k_gemm<<<dim3(16, 128), 256, 0, stream>>>(hall, wkb, BB * 1024, HD, HD, 0, 1,
                                            1.0f, nullptr, kbuf);
  k_gemm<<<dim3(16, 128), 256, 0, stream>>>(hall, wvb, BB * 1024, HD, HD, 0, 1,
                                            1.0f, nullptr, vbuf);

  // ---- banded attention + output projection + LN1
  k_attn<<<dim3(BB * NH, MM / 16), 256, 0, stream>>>(qb, kbuf, vbuf, kpeb, ctxb);
  k_gemm<<<dim3(16, 64), 256, 0, stream>>>(ctxb, wob, NTOK, HD, HD, 0, 0,
                                           1.0f, attno, nullptr);
  k_ln1<<<NTOK, 256, 0, stream>>>(h, attno, ln1_g, ln1_b, h1f, h1b);

  // ---- MoE: gate -> compact -> grouped GEMMs -> combine + LN2
  k_zero_counts<<<1, 32, 0, stream>>>(count);
  k_gate<<<NTOK, 32, 0, stream>>>(h1f, gate_w, gate_b, topi, topw, count);
  k_scan<<<1, 32, 0, stream>>>(count, base, fill);
  k_assign<<<16, 256, 0, stream>>>(topi, base, fill, slot_token, inv_slot);
  k_moe1<<<dim3(DFF / 64, NEXP, 64), 256, 0, stream>>>(h1b, w1b, b1, base, count,
                                                       slot_token, y1);
  k_moe2<<<dim3(HD / 64, NEXP, 64), 256, 0, stream>>>(y1, w2b, b2, base, count, y2);
  k_ln2<<<NTOK, 256, 0, stream>>>(h1f, y2, inv_slot, topw, ln2_g, ln2_b, out);
}
